// LSTMDecoder_78984448573629
// MI455X (gfx1250) — matrix operand order repaired and checked by reading, not compiled, not run on hardware
//
#include <hip/hip_runtime.h>
#include <hip/hip_bf16.h>
#include <math.h>

// ---------------------------------------------------------------------------
// Problem constants (from reference)
// ---------------------------------------------------------------------------
#define VOCAB 32000
#define EMB   256
#define HID   512
#define FEAT  1024
#define BSZ   32
#define TLEN  100

// ---------------------------------------------------------------------------
// CDNA5 WMMA types
// ---------------------------------------------------------------------------
typedef __attribute__((ext_vector_type(16))) __bf16     v16bf;
typedef __attribute__((ext_vector_type(8)))  float      v8f;
typedef __attribute__((ext_vector_type(4)))  unsigned   u32x4;
typedef __attribute__((ext_vector_type(4)))  int        v4i;

union ABu { v16bf v; u32x4 q[2]; };

// Optional CDNA5 async global->LDS copy (ASYNCcnt-tracked). Guarded so the
// file compiles on toolchains that don't declare the builtins.
#if defined(__gfx1250__) && \
    __has_builtin(__builtin_amdgcn_global_load_async_to_lds_b128) && \
    __has_builtin(__builtin_amdgcn_s_wait_asynccnt)
#define HAVE_ASYNC_LDS 1
typedef __attribute__((address_space(1))) v4i* gv4i_p;   // global v4i*
typedef __attribute__((address_space(3))) v4i* lv4i_p;   // LDS v4i*
#else
#define HAVE_ASYNC_LDS 0
#endif

// float -> bf16 round-to-nearest-even
__device__ __forceinline__ unsigned short f2bf(float f) {
    unsigned u = __float_as_uint(f);
    unsigned r = u + 0x7FFFu + ((u >> 16) & 1u);
    return (unsigned short)(r >> 16);
}

// A operand, 16x32 bf16 tile (MxK), memory row-major with leading dim `ld`
// (elements). Per ISA 7.12.2: lanes 0-15 hold M=lane, K=0..7 & 16..23;
// lanes 16-31 hold M=lane-16, K=8..15 & 24..31.
__device__ __forceinline__ v16bf loadA16x32(const unsigned short* base, int ld, int lane) {
    int l  = lane & 15;
    int hi = lane >> 4;               // 0 or 1
    const unsigned short* p = base + l * ld + hi * 8;
    ABu a;
    a.q[0] = *(const u32x4*)(p);       // K = hi*8 .. hi*8+7
    a.q[1] = *(const u32x4*)(p + 16);  // K = hi*8+16 .. hi*8+23
    return a.v;
}

// B operand, 32x16 bf16 tile (KxN) where memory holds W as (N,K) row-major,
// i.e. B = W^T. Matches A: lanes 0-15 hold N=lane, K=0..7 & 16..23; lanes 16-31 hold
// N=lane-16, K=8..15 & 24..31.
__device__ __forceinline__ v16bf loadB16x32(const unsigned short* base, int ld, int lane) {
    int l  = lane & 15;
    int hi = lane >> 4;
    const unsigned short* p = base + l * ld + hi * 8;
    ABu b;
    b.q[0] = *(const u32x4*)(p);
    b.q[1] = *(const u32x4*)(p + 16);
    return b.v;
}

__device__ __forceinline__ v8f wmma_bf16(v16bf a, v16bf b, v8f c) {
    return __builtin_amdgcn_wmma_f32_16x16x32_bf16(false, a, false, b, (short)0, c, false, false);
}

// ---------------------------------------------------------------------------
// K0: plain f32 -> bf16 conversion (Wv)
// ---------------------------------------------------------------------------
__global__ void k_convert(const float* __restrict__ src, unsigned short* __restrict__ dst, int n) {
    int i = blockIdx.x * blockDim.x + threadIdx.x;
    if (i < n) dst[i] = f2bf(src[i]);
}

// ---------------------------------------------------------------------------
// K1: gate-permuting weight conversion.  dst row np = 4*h + g  <-  src row
// g*512 + h  (torch gate order i,f,g,o).  kshift = log2(K).
// ---------------------------------------------------------------------------
__global__ void k_permute_w(const float* __restrict__ src, unsigned short* __restrict__ dst,
                            int kshift, int n) {
    int i = blockIdx.x * blockDim.x + threadIdx.x;
    if (i >= n) return;
    int K  = 1 << kshift;
    int np = i >> kshift;
    int k  = i & (K - 1);
    int h  = np >> 2;
    int g  = np & 3;
    dst[i] = f2bf(src[(g * HID + h) * K + k]);
}

__global__ void k_bias(const float* __restrict__ bih, const float* __restrict__ bhh,
                       float* __restrict__ pb) {
    int np = blockIdx.x * blockDim.x + threadIdx.x;
    if (np >= 4 * HID) return;
    int h = np >> 2, g = np & 3;
    pb[np] = bih[g * HID + h] + bhh[g * HID + h];
}

// ---------------------------------------------------------------------------
// K2: embedding gather + bf16 convert.  X[t][b][e], teacher forcing:
// tok(t=0)=START(1), else reports[b][t-1].
// ---------------------------------------------------------------------------
__global__ void k_embed(const float* __restrict__ emb, const int* __restrict__ reports,
                        unsigned short* __restrict__ X, int n) {
    int i = blockIdx.x * blockDim.x + threadIdx.x;
    if (i >= n) return;
    int e = i & (EMB - 1);
    int b = (i >> 8) & (BSZ - 1);
    int t = i >> 13;                       // /(32*256)
    int tok = (t == 0) ? 1 : reports[b * TLEN + (t - 1)];
    X[i] = f2bf(emb[tok * EMB + e]);
}

// ---------------------------------------------------------------------------
// K3: avg-pool(7x7) + FC -> h0 (f32).  One block per batch row.
// ---------------------------------------------------------------------------
__global__ __launch_bounds__(256) void k_poolfc(const float* __restrict__ features,
                                                const float* __restrict__ fcw,
                                                const float* __restrict__ fcb,
                                                float* __restrict__ h0) {
    __shared__ float pooled[FEAT];
    int b = blockIdx.x, tid = threadIdx.x;
    for (int f = tid; f < FEAT; f += 256) {
        const float* p = features + (b * FEAT + f) * 49;
        float s = 0.f;
        #pragma unroll
        for (int j = 0; j < 49; ++j) s += p[j];
        pooled[f] = s * (1.0f / 49.0f);
    }
    __syncthreads();
    for (int h = tid; h < HID; h += 256) {
        const float* wr = fcw + h * FEAT;
        float s = fcb[h];
        for (int f = 0; f < FEAT; ++f) s += pooled[f] * wr[f];
        h0[b * HID + h] = s;
    }
}

// ---------------------------------------------------------------------------
// K4: sequential LSTM, single persistent workgroup, 16 waves.
// Per step: gates(32x2048) = X_t @ Wih^T + h @ Whh^T + pb via bf16 WMMA.
// Gate columns are pre-permuted (4h+g) so each 16-wide N tile is wave-local
// for the cell update.  h,c state live in LDS; h_t saved (bf16) to H.
// ---------------------------------------------------------------------------
__global__ __launch_bounds__(512) void k_lstm(const unsigned short* __restrict__ X,
                                              const unsigned short* __restrict__ Wih,
                                              const unsigned short* __restrict__ Whh,
                                              const float* __restrict__ pb,
                                              const float* __restrict__ h0,
                                              unsigned short* __restrict__ H) {
    __shared__ unsigned short hprev[2][BSZ * HID];  // 2 x 32 KB ping-pong (bf16)
    __shared__ float          cbuf[BSZ * HID];      // 64 KB cell state
    __shared__ float          dstage[16][256];      // per-wave 16x16 gate tile

    int tid = threadIdx.x, lane = tid & 31, w = tid >> 5;

    for (int i = tid; i < BSZ * HID; i += 512) {
        hprev[0][i] = f2bf(h0[i]);
        cbuf[i] = 0.f;
    }
    __syncthreads();

    int cur = 0;
    for (int t = 0; t < TLEN; ++t) {
        const unsigned short* Xt = X + t * (BSZ * EMB);
        #pragma unroll 1
        for (int i = 0; i < 16; ++i) {            // 16 of 256 gate tiles per wave
            int q  = (w << 4) | i;
            int mt = q >> 7;                      // 0..1  (batch tile)
            int nt = q & 127;                     // 0..127 (gate-col tile)
            int col = (nt << 4) + (lane & 15);
            float bsc = pb[col];
            v8f acc = {bsc, bsc, bsc, bsc, bsc, bsc, bsc, bsc};

            const unsigned short* Ax = Xt  + (mt << 4) * EMB;
            const unsigned short* Bx = Wih + (nt << 4) * EMB;
            #pragma unroll 2
            for (int kk = 0; kk < 8; ++kk)        // K = 256
                acc = wmma_bf16(loadA16x32(Ax + kk * 32, EMB, lane),
                                loadB16x32(Bx + kk * 32, EMB, lane), acc);

            const unsigned short* Ah = &hprev[cur][(mt << 4) * HID];
            const unsigned short* Bh = Whh + (nt << 4) * HID;
            #pragma unroll 2
            for (int kk = 0; kk < 16; ++kk)       // K = 512
                acc = wmma_bf16(loadA16x32(Ah + kk * 32, HID, lane),
                                loadB16x32(Bh + kk * 32, HID, lane), acc);

            // stage D tile (C layout: vgpr r -> M=r / r+8, lane -> N)
            #pragma unroll
            for (int r = 0; r < 8; ++r) {
                int m = r + ((lane >> 4) << 3);
                dstage[w][m * 16 + (lane & 15)] = acc[r];
            }
            __builtin_amdgcn_wave_barrier();

            // cell update: 64 (m,h') pairs per tile, 2 per lane
            #pragma unroll
            for (int p = 0; p < 2; ++p) {
                int pi = (lane << 1) | p;
                int m  = pi & 15;
                int hq = pi >> 4;                 // 0..3
                float gi = dstage[w][m * 16 + hq * 4 + 0];
                float gf = dstage[w][m * 16 + hq * 4 + 1];
                float gg = dstage[w][m * 16 + hq * 4 + 2];
                float go = dstage[w][m * 16 + hq * 4 + 3];
                int bidx = (mt << 4) + m;
                int hidx = (nt << 2) + hq;
                int ci   = bidx * HID + hidx;
                float c  = cbuf[ci];
                float iv = 1.f / (1.f + __expf(-gi));
                float fv = 1.f / (1.f + __expf(-gf));
                float gv = tanhf(gg);
                float ov = 1.f / (1.f + __expf(-go));
                c = fv * c + iv * gv;
                float hv = ov * tanhf(c);
                cbuf[ci] = c;
                unsigned short hb = f2bf(hv);
                hprev[cur ^ 1][ci] = hb;
                H[t * (BSZ * HID) + ci] = hb;
            }
            __builtin_amdgcn_wave_barrier();
        }
        __syncthreads();
        cur ^= 1;
    }
}

// ---------------------------------------------------------------------------
// K5: logits GEMM.  M = T*B = 3200 (H bf16, K=512 row-major), N = 32000
// (Wv bf16, (N,K) row-major, L2-resident), K = 512.
// Block = 8 waves computing a 64(M) x 128(N) macro-tile: the 64x512 bf16
// A-slab (64 KB) is staged in LDS (async global->LDS when available), each
// wave owns one 16-wide N tile and register-blocks 4 M tiles so every B
// fragment feeds 4 WMMAs (4x Wv traffic reuse).
// ---------------------------------------------------------------------------
__global__ __launch_bounds__(256) void k_logits(const unsigned short* __restrict__ H,
                                                const unsigned short* __restrict__ Wv,
                                                const float* __restrict__ bv,
                                                float* __restrict__ out) {
    __shared__ unsigned short sA[64 * HID];       // 64 KB
    int tid = threadIdx.x;
    int mt  = blockIdx.y;                         // 64-row macro tile index

    // stage A-slab: 64 rows x 512 bf16 = 4096 x 16B chunks
    const unsigned short* gA = H + mt * 64 * HID;
#if HAVE_ASYNC_LDS
    for (int j = tid; j < 4096; j += 256) {
        __builtin_amdgcn_global_load_async_to_lds_b128(
            (gv4i_p)(unsigned short*)(gA + j * 8),
            (lv4i_p)(&sA[j * 8]),
            0, 0);
    }
    __builtin_amdgcn_s_wait_asynccnt(0);
#else
    {
        const u32x4* src = (const u32x4*)gA;
        u32x4*       dst = (u32x4*)sA;
        for (int j = tid; j < 4096; j += 256) dst[j] = src[j];
    }
#endif
    __syncthreads();

    int w = tid >> 5, lane = tid & 31;
    int nt = blockIdx.x * 8 + w;
    const unsigned short* Bbase = Wv + nt * 16 * HID;

    v8f acc[4];
    #pragma unroll
    for (int mi = 0; mi < 4; ++mi)
        acc[mi] = (v8f){0.f, 0.f, 0.f, 0.f, 0.f, 0.f, 0.f, 0.f};

    #pragma unroll 2
    for (int kk = 0; kk < 16; ++kk) {
        // prefetch next-k Wv fragment for this lane (global_prefetch path)
        __builtin_prefetch(Bbase + ((kk + 1) & 15) * 32 + (lane & 15) * HID + (lane >> 4) * 16, 0, 0);
        v16bf b = loadB16x32(Bbase + kk * 32, HID, lane);
        #pragma unroll
        for (int mi = 0; mi < 4; ++mi)
            acc[mi] = wmma_bf16(loadA16x32(sA + (mi * 16) * HID + kk * 32, HID, lane), b, acc[mi]);
    }

    int col = (nt << 4) + (lane & 15);
    float bias = bv[col];
    #pragma unroll
    for (int mi = 0; mi < 4; ++mi) {
        #pragma unroll
        for (int r = 0; r < 8; ++r) {
            int m   = r + ((lane >> 4) << 3);
            int row = (mt << 6) + (mi << 4) + m;  // row = t*32 + b
            int tt  = row >> 5;
            int bb  = row & 31;
            out[(bb * TLEN + tt) * VOCAB + col] = acc[mi][r] + bias;
        }
    }
}

// ---------------------------------------------------------------------------
// Host-side launch
// ---------------------------------------------------------------------------
extern "C" void kernel_launch(void* const* d_in, const int* in_sizes, int n_in,
                              void* d_out, int out_size, void* d_ws, size_t ws_size,
                              hipStream_t stream) {
    (void)in_sizes; (void)n_in; (void)out_size; (void)ws_size;

    const float* features = (const float*)d_in[0];
    const int*   reports  = (const int*)  d_in[1];
    const float* emb      = (const float*)d_in[2];
    const float* fcw      = (const float*)d_in[3];
    const float* fcb      = (const float*)d_in[4];
    const float* W_ih     = (const float*)d_in[5];
    const float* b_ih     = (const float*)d_in[6];
    const float* W_hh     = (const float*)d_in[7];
    const float* b_hh     = (const float*)d_in[8];
    const float* Wv       = (const float*)d_in[9];
    const float* bv       = (const float*)d_in[10];
    float* out = (float*)d_out;

    // workspace layout (all offsets 16B aligned)
    char* wsp = (char*)d_ws;
    float*          h0   = (float*)         (wsp);                 //   65536 B
    unsigned short* X    = (unsigned short*)(wsp + 65536);         // 1638400 B
    unsigned short* Wihb = (unsigned short*)(wsp + 1703936);       // 1048576 B
    unsigned short* Whhb = (unsigned short*)(wsp + 2752512);       // 2097152 B
    unsigned short* Wvb  = (unsigned short*)(wsp + 4849664);       // 32768000 B
    float*          pb   = (float*)         (wsp + 37617664);      //    8192 B
    unsigned short* Hb   = (unsigned short*)(wsp + 37625856);      // 3276800 B

    int nWv  = VOCAB * HID;          // 16,384,000
    int nWih = 4 * HID * EMB;        //    524,288
    int nWhh = 4 * HID * HID;        //  1,048,576
    int nX   = TLEN * BSZ * EMB;     //    819,200

    k_convert  <<<(nWv  + 255) / 256, 256, 0, stream>>>(Wv, Wvb, nWv);
    k_permute_w<<<(nWih + 255) / 256, 256, 0, stream>>>(W_ih, Wihb, 8, nWih);
    k_permute_w<<<(nWhh + 255) / 256, 256, 0, stream>>>(W_hh, Whhb, 9, nWhh);
    k_bias     <<<(4 * HID + 255) / 256, 256, 0, stream>>>(b_ih, b_hh, pb);
    k_embed    <<<(nX   + 255) / 256, 256, 0, stream>>>(emb, reports, X, nX);
    k_poolfc   <<<BSZ, 256, 0, stream>>>(features, fcw, fcb, h0);
    k_lstm     <<<1, 512, 0, stream>>>(X, Wihb, Whhb, pb, h0, Hb);
    k_logits   <<<dim3(VOCAB / (16 * 8), (TLEN * BSZ) / 64), 256, 0, stream>>>(Hb, Wvb, bv, out);
}
